// Conv2dNotOpti_4604204941396
// MI455X (gfx1250) — matrix operand order repaired and checked by reading, not compiled, not run on hardware
//
#include <hip/hip_runtime.h>
#include <hip/hip_bf16.h>

typedef __attribute__((ext_vector_type(16))) _Float16 v16h;
typedef __attribute__((ext_vector_type(8)))  _Float16 v8h;
typedef __attribute__((ext_vector_type(8)))  float    v8f;

constexpr int NB    = 32;
constexpr int CIN   = 128;
constexpr int COUT  = 256;
constexpr int H     = 56;
constexpr int W     = 56;
constexpr int HW    = H * W;            // 3136
constexpr int KRED  = CIN * 9;          // 1152
constexpr int M_TILE = 128;
constexpr int N_TILE = 128;
constexpr int K_STEP = 32;
constexpr int KSTEPS = KRED / K_STEP;   // 36 = 9 (rs) * 4 (c-blocks of 32)
constexpr int LDSS   = 40;              // half-stride per row: 80B, keeps 16B alignment

__global__ __launch_bounds__(256)
void conv3x3_implicit_gemm_wmma(const float* __restrict__ X,
                                const float* __restrict__ Wt,
                                const float* __restrict__ bias,
                                float* __restrict__ out)
{
    // Double-buffered staging; K axis is PERMUTED to k' = rs*128 + c on both A and B.
    __shared__ __align__(16) _Float16 As[2][M_TILE][LDSS]; // weights [m][k']
    __shared__ __align__(16) _Float16 Bs[2][N_TILE][LDSS]; // im2col  [n][k'] (col-major B)

    const int tid   = threadIdx.x;
    const int lane  = tid & 31;
    const int wave  = tid >> 5;
    const int waveM = wave & 3;       // 4 waves x 32 rows
    const int waveN = wave >> 2;      // 2 waves x 64 cols
    const int mBase = blockIdx.y * M_TILE;
    const int jBase = blockIdx.x * N_TILE;

    // Staging ownership: thread owns LDS row (tcol) and 16 consecutive k' (tsel*16).
    const int tcol = tid & 127;
    const int tsel = tid >> 7;        // 0..1

    // Per-thread hoisted im2col geometry for B staging.
    const int jS  = jBase + tcol;
    const int nS  = jS / HW;
    const int hwS = jS - nS * HW;
    const int ohS = hwS / W;
    const int owS = hwS - ohS * W;
    const float* bBase = X + (size_t)nS * CIN * HW + (size_t)(tsel * 16) * HW;
    const float* aBase = Wt + (size_t)(mBase + tcol) * KRED + (size_t)(tsel * 16) * 9;

    // Fragment-load lane constants (CDNA5 16-bit A/B VGPR layouts).
    const int laneLo = lane & 15;
    const int kh0 = (lane < 16) ? 0 : 8;   // A: K {0..7,16..23} vs {8..15,24..31}
    const int kb0 = (lane < 16) ? 0 : 8;   // B: K {0..7,16..23} vs {8..15,24..31}

    v8f acc[2][4] = {};

    auto stage = [&](int step, int buf) {
        const int rs = step >> 2;              // scalar per step
        const int c0 = (step & 3) * 32;        // scalar per step
        const int r  = rs / 3;
        const int s  = rs - 3 * r;
        // ---- B: im2col gather. Clamp the address so ALL 16 loads are unconditional
        //      (one clause, no exec toggling); zero the halo afterwards with cndmasks.
        const int ih = ohS + r - 1;
        const int iw = owS + s - 1;
        const bool ok  = ((unsigned)ih < (unsigned)H) && ((unsigned)iw < (unsigned)W);
        const int  ihc = ih < 0 ? 0 : (ih > H - 1 ? H - 1 : ih);
        const int  iwc = iw < 0 ? 0 : (iw > W - 1 ? W - 1 : iw);
        const float* bp = bBase + c0 * HW + ihc * W + iwc;
        v8h b0, b1;
        #pragma unroll
        for (int q = 0; q < 8; ++q) {
            b0[q] = (_Float16)bp[q * HW];
            b1[q] = (_Float16)bp[(q + 8) * HW];
        }
        if (!ok) {
            #pragma unroll
            for (int q = 0; q < 8; ++q) { b0[q] = (_Float16)0.0f; b1[q] = (_Float16)0.0f; }
        }
        *(v8h*)(&Bs[buf][tcol][tsel * 16])     = b0;
        *(v8h*)(&Bs[buf][tcol][tsel * 16 + 8]) = b1;
        // ---- A: weights, stride-9 gather (L2-resident), k' = rs*128 + c ----
        const float* ap = aBase + c0 * 9 + rs;
        v8h a0, a1;
        #pragma unroll
        for (int q = 0; q < 8; ++q) {
            a0[q] = (_Float16)ap[q * 9];
            a1[q] = (_Float16)ap[(q + 8) * 9];
        }
        *(v8h*)(&As[buf][tcol][tsel * 16])     = a0;
        *(v8h*)(&As[buf][tcol][tsel * 16 + 8]) = a1;
    };

    auto compute = [&](int buf) {
        v16h afrag[2];
        #pragma unroll
        for (int mt = 0; mt < 2; ++mt) {
            int row = waveM * 32 + mt * 16 + laneLo;
            v8h lo = *(const v8h*)(&As[buf][row][kh0]);
            v8h hi = *(const v8h*)(&As[buf][row][kh0 + 16]);
            #pragma unroll
            for (int e = 0; e < 8; ++e) { afrag[mt][e] = lo[e]; afrag[mt][e + 8] = hi[e]; }
        }
        #pragma unroll
        for (int nt = 0; nt < 4; ++nt) {
            int col = waveN * 64 + nt * 16 + laneLo;
            v8h lo = *(const v8h*)(&Bs[buf][col][kb0]);
            v8h hi = *(const v8h*)(&Bs[buf][col][kb0 + 16]);
            v16h bfrag;
            #pragma unroll
            for (int e = 0; e < 8; ++e) { bfrag[e] = lo[e]; bfrag[e + 8] = hi[e]; }
            #pragma unroll
            for (int mt = 0; mt < 2; ++mt)
                acc[mt][nt] = __builtin_amdgcn_wmma_f32_16x16x32_f16(
                    false, afrag[mt], false, bfrag,
                    (short)0, acc[mt][nt], false, false);
        }
    };

    // ---- Pipelined main loop: stage(next) overlaps compute(cur), 1 barrier/iter ----
    stage(0, 0);
    __syncthreads();
    for (int step = 0; step < KSTEPS; ++step) {
        const int buf = step & 1;
        if (step + 1 < KSTEPS) stage(step + 1, buf ^ 1);
        compute(buf);
        __syncthreads();
    }

    // ---- Epilogue: D layout (VGPR r: M = r + 8*(lane>=16), N = lane&15), bias, NT stores ----
    const int laneHi = lane >> 4;
    float bv[2][8];
    #pragma unroll
    for (int mt = 0; mt < 2; ++mt)
        #pragma unroll
        for (int rr = 0; rr < 8; ++rr)
            bv[mt][rr] = bias[mBase + waveM * 32 + mt * 16 + laneHi * 8 + rr];

    #pragma unroll
    for (int mt = 0; mt < 2; ++mt) {
        #pragma unroll
        for (int nt = 0; nt < 4; ++nt) {
            int jg = jBase + waveN * 64 + nt * 16 + laneLo;
            int n  = jg / HW;
            int hw = jg - n * HW;
            float* op = out + (size_t)n * COUT * HW + hw;
            #pragma unroll
            for (int rr = 0; rr < 8; ++rr) {
                int k = mBase + waveM * 32 + mt * 16 + laneHi * 8 + rr;
                __builtin_nontemporal_store(acc[mt][nt][rr] + bv[mt][rr], op + (size_t)k * HW);
            }
        }
    }
}

extern "C" void kernel_launch(void* const* d_in, const int* in_sizes, int n_in,
                              void* d_out, int out_size, void* d_ws, size_t ws_size,
                              hipStream_t stream) {
    (void)in_sizes; (void)n_in; (void)out_size; (void)d_ws; (void)ws_size;
    const float* X    = (const float*)d_in[0];
    const float* Wt   = (const float*)d_in[1];
    const float* bias = (const float*)d_in[2];
    float* out        = (float*)d_out;

    dim3 grid((NB * HW) / N_TILE, COUT / M_TILE); // (784, 2)
    dim3 block(256);
    conv3x3_implicit_gemm_wmma<<<grid, block, 0, stream>>>(X, Wt, bias, out);
}
